// BiLSTM_CRF_43044162241186
// MI455X (gfx1250) — compile-verified, matrix operand order checked by reading, not run on hardware
//
#include <hip/hip_runtime.h>
#include <hip/hip_bf16.h>

// ---------------- problem constants ----------------
#define Bsz   128
#define Tlen  256
#define Edim  300
#define EPad  320          // 300 padded to multiple of 32 for WMMA K
#define Hdim  256
#define H2    512          // 2*H (bi-directional concat)
#define G4    1024         // 4*H gate width
#define Ktag  12
#define MTOT  (Bsz * Tlen) // 32768 GEMM rows

typedef __attribute__((ext_vector_type(16))) __bf16 v16bf;
typedef __attribute__((ext_vector_type(8)))  float  v8f;

__device__ __forceinline__ v8f wmma_bf16(v16bf a, v16bf b, v8f c) {
  // v_wmma_f32_16x16x32_bf16  (neg_a, A, neg_b, B, c_mod, C, reuse_a, reuse_b)
  return __builtin_amdgcn_wmma_f32_16x16x32_bf16(false, a, false, b, (short)0, c, false, false);
}

// Load one 16-lane-row fragment (A or B) for 16x16x32 bf16 WMMA.
// p = base of this lane's row (row-major, contiguous in K); kbase already
// includes k0 + (lane>=16 ? 8 : 0). Per ISA 7.12.2: elements 0..7 -> K=kbase..kbase+7,
// elements 8..15 -> K=kbase+16..kbase+23 (two contiguous 16B chunks).
__device__ __forceinline__ v16bf frag_ld(const __bf16* __restrict__ p, int kbase) {
  v16bf f;
#pragma unroll
  for (int e = 0; e < 8; ++e) f[e] = p[kbase + e];
#pragma unroll
  for (int e = 0; e < 8; ++e) f[8 + e] = p[kbase + 16 + e];
  return f;
}

// ---- CDNA5 async LDS<->global DMA (ASYNCcnt-tracked), gfx1250 VFLAT forms ----
__device__ __forceinline__ void async_ld_b128(void* lds_dst, const void* gsrc) {
  unsigned int l = (unsigned int)(unsigned long long)lds_dst; // addr[31:0] = LDS offset
  asm volatile("global_load_async_to_lds_b128 %0, %1, off" ::"v"(l), "v"(gsrc)
               : "memory");
}
__device__ __forceinline__ void async_st_b128(void* gdst, const void* lds_src) {
  unsigned int l = (unsigned int)(unsigned long long)lds_src;
  asm volatile("global_store_async_from_lds_b128 %0, %1, off" ::"v"(gdst), "v"(l)
               : "memory");
}
__device__ __forceinline__ void wait_async0() {
  asm volatile("s_wait_asynccnt 0" ::: "memory");
}

__device__ __forceinline__ float sigmoidf(float x) { return 1.0f / (1.0f + __expf(-x)); }

// ---------------- prep kernels ----------------
__global__ __launch_bounds__(256) void k_prep_wih(const float* __restrict__ wf,
                                                  const float* __restrict__ wb,
                                                  __bf16* __restrict__ out) {
  int idx = blockIdx.x * 256 + threadIdx.x;
  int total = 2 * G4 * EPad;
  if (idx >= total) return;
  int dir = idx / (G4 * EPad);
  int rem = idx - dir * (G4 * EPad);
  int n = rem / EPad, k = rem - n * EPad;
  const float* src = dir ? wb : wf;
  float v = (k < Edim) ? src[n * Edim + k] : 0.0f;
  out[idx] = (__bf16)v;
}

__global__ __launch_bounds__(256) void k_prep_whh(const float* __restrict__ wf,
                                                  const float* __restrict__ wb,
                                                  __bf16* __restrict__ out) {
  int idx = blockIdx.x * 256 + threadIdx.x;
  int total = 2 * G4 * Hdim;
  if (idx >= total) return;
  int dir = idx / (G4 * Hdim);
  int rem = idx - dir * (G4 * Hdim);
  const float* src = dir ? wb : wf;
  out[idx] = (__bf16)src[rem];
}

// h2t_w (12 x 512) -> bf16 padded to (16 x 512); rows 12..15 zero.
__global__ __launch_bounds__(256) void k_prep_h2t(const float* __restrict__ w,
                                                  __bf16* __restrict__ out) {
  int idx = blockIdx.x * 256 + threadIdx.x;
  int total = 16 * H2;
  if (idx >= total) return;
  int n = idx / H2, k = idx - n * H2;
  out[idx] = (__bf16)((n < Ktag) ? w[n * H2 + k] : 0.0f);
}

// Gather embedding rows for every (b,t) token into bf16 padded K=320.
__global__ __launch_bounds__(256) void k_gather(const int* __restrict__ sentences,
                                                const float* __restrict__ embed,
                                                __bf16* __restrict__ xBF) {
  int idx = blockIdx.x * 256 + threadIdx.x;
  int total = MTOT * EPad;
  if (idx >= total) return;
  int m = idx / EPad, k = idx - m * EPad;
  float v = 0.0f;
  if (k < Edim) {
    int tok = sentences[m];
    v = embed[(size_t)tok * Edim + k];
  }
  xBF[idx] = (__bf16)v;
}

// ---------------- input GEMM: gatesIn = x @ W_ih^T  (M=32768, K=320, N=2048) ----------------
__global__ __launch_bounds__(128) void k_ingemm(const __bf16* __restrict__ xBF,
                                                const __bf16* __restrict__ wihBF,
                                                __bf16* __restrict__ gatesIn) {
  int lane = threadIdx.x & 31;
  int wave = threadIdx.x >> 5;
  int r = lane & 15;
  int khalf = (lane & 16) ? 8 : 0;
  int m0 = blockIdx.x * 16;
  int n0 = blockIdx.y * 256 + wave * 64;

  v8f acc[4] = {};
  const __bf16* arow = xBF + (size_t)(m0 + r) * EPad;

#pragma unroll 1
  for (int k0 = 0; k0 < EPad; k0 += 32) {
    v16bf a = frag_ld(arow, k0 + khalf);
#pragma unroll
    for (int s = 0; s < 4; ++s) {
      int n = n0 + s * 16 + r;
      v16bf b = frag_ld(wihBF + (size_t)n * EPad, k0 + khalf);
      acc[s] = wmma_bf16(a, b, acc[s]);
    }
  }

#pragma unroll
  for (int s = 0; s < 4; ++s) {
    int n = n0 + s * 16 + r;
    int dir = n >> 10;
    int nloc = n & 1023;
#pragma unroll
    for (int v = 0; v < 8; ++v) {
      int m = m0 + v + khalf;
      gatesIn[(size_t)dir * MTOT * G4 + (size_t)m * G4 + nloc] = (__bf16)acc[s][v];
    }
  }
}

// ---------------- LSTM recurrence ----------------
// grid (8,2): x = 16-row batch tile, y = direction. 512 threads = 16 waves.
// Per step: async-DMA the 16x1024 bf16 gatesIn tile straight into LDS (overlapped
// with the h@W_hh^T WMMA loop), RMW with accumulators+bias, elementwise cell,
// then async-store the new h tile to lstm_out from LDS (overlapped with next step).
__global__ __launch_bounds__(512) void k_lstm(const __bf16* __restrict__ gatesIn,
                                              const __bf16* __restrict__ whhBF,
                                              const float* __restrict__ bihf,
                                              const float* __restrict__ bhhf,
                                              const float* __restrict__ bihb,
                                              const float* __restrict__ bhhb,
                                              __bf16* __restrict__ lstmOut) {
  __shared__ __bf16 hBF[16 * Hdim];   // 8 KB   current hidden state (WMMA A operand)
  __shared__ float  cS[16 * Hdim];    // 16 KB  cell state
  __shared__ __bf16 gateS[16 * G4];   // 32 KB  gate tile (async-load target, then RMW)
  __shared__ float  bsum[G4];         // 4 KB   b_ih + b_hh

  int dir = blockIdx.y;
  int b0 = blockIdx.x * 16;
  const float* bih = dir ? bihb : bihf;
  const float* bhh = dir ? bhhb : bhhf;

  for (int i = threadIdx.x; i < G4; i += 512) bsum[i] = bih[i] + bhh[i];
  for (int i = threadIdx.x; i < 16 * Hdim; i += 512) {
    hBF[i] = (__bf16)0.0f;
    cS[i] = 0.0f;
  }
  __syncthreads();

  int lane = threadIdx.x & 31;
  int wave = threadIdx.x >> 5;    // 0..15
  int r = lane & 15;
  int khalf = (lane & 16) ? 8 : 0;

  const __bf16* whh = whhBF + (size_t)dir * G4 * Hdim;
  const __bf16* gIn = gatesIn + (size_t)dir * MTOT * G4;

  for (int s = 0; s < Tlen; ++s) {
    int t = dir ? (Tlen - 1 - s) : s;

    // (a) kick off async DMA of this step's gate tile: wave handles row `wave`
    {
      const char* gsrc = (const char*)(gIn + ((size_t)(b0 + wave) * Tlen + t) * G4);
      char* ldst = (char*)gateS + wave * (G4 * 2);
#pragma unroll
      for (int i = 0; i < 4; ++i) {
        int chunk = (i * 32 + lane) * 16;   // 2048 B per row = 4 issues x 32 lanes x 16 B
        async_ld_b128(ldst + chunk, gsrc + chunk);
      }
    }

    // (b) h @ W_hh^T : this wave owns gate columns [wave*64, wave*64+64)
    v8f acc[4] = {};
#pragma unroll 1
    for (int k0 = 0; k0 < Hdim; k0 += 32) {
      v16bf a = frag_ld(&hBF[r * Hdim], k0 + khalf);
#pragma unroll
      for (int nt = 0; nt < 4; ++nt) {
        int n = wave * 64 + nt * 16 + r;
        const __bf16* brow = whh + (size_t)n * Hdim;
        __builtin_prefetch((const void*)(brow + k0 + 32), 0, 1); // stream W_hh via L2
        v16bf b = frag_ld(brow, k0 + khalf);
        acc[nt] = wmma_bf16(a, b, acc[nt]);
      }
    }

    // (c) DMA (this step's loads + last step's h store) must have landed
    wait_async0();
    __syncthreads();

    // (d) gates = inputProj (in LDS) + bias + recurrent accumulators  (in-place RMW)
#pragma unroll
    for (int nt = 0; nt < 4; ++nt) {
      int n = wave * 64 + nt * 16 + r;
#pragma unroll
      for (int v = 0; v < 8; ++v) {
        int m = v + khalf;
        int idx = m * G4 + n;
        gateS[idx] = (__bf16)(acc[nt][v] + bsum[n] + (float)gateS[idx]);
      }
    }
    __syncthreads();

    // (e) elementwise LSTM cell
    for (int p = threadIdx.x; p < 16 * Hdim; p += 512) {
      int m = p >> 8;
      int j = p & 255;
      float ig = sigmoidf((float)gateS[m * G4 + j]);
      float fg = sigmoidf((float)gateS[m * G4 + 256 + j]);
      float gg = tanhf((float)gateS[m * G4 + 512 + j]);
      float og = sigmoidf((float)gateS[m * G4 + 768 + j]);
      float c = fg * cS[p] + ig * gg;
      cS[p] = c;
      hBF[p] = (__bf16)(og * tanhf(c));
    }
    __syncthreads();

    // (f) async-store h row `wave` (256 bf16 = 512 B = one 32-lane b128 issue)
    {
      char* gdst = (char*)(lstmOut + ((size_t)(b0 + wave) * Tlen + t) * H2 + dir * Hdim);
      const char* lsrc = (const char*)hBF + wave * (Hdim * 2);
      async_st_b128(gdst + lane * 16, lsrc + lane * 16);
    }
    // no extra barrier: next step's DMA/gateS writes are fenced by the
    // post-elementwise barrier; GEMM & async store only read hBF.
  }
  wait_async0();
}

// ---------------- feats GEMM: feats = lstm_out @ h2t^T + b  (M=32768, K=512, N=16) ----------------
__global__ __launch_bounds__(128) void k_featgemm(const __bf16* __restrict__ lstmOut,
                                                  const __bf16* __restrict__ h2tBF,
                                                  const float* __restrict__ h2t_b,
                                                  float* __restrict__ feats) {
  int lane = threadIdx.x & 31;
  int wave = threadIdx.x >> 5;
  int r = lane & 15;
  int khalf = (lane & 16) ? 8 : 0;
  int m0 = (blockIdx.x * 4 + wave) * 16;

  v8f acc = {};
  const __bf16* arow = lstmOut + (size_t)(m0 + r) * H2;
  const __bf16* brow = h2tBF + (size_t)r * H2;

#pragma unroll 1
  for (int k0 = 0; k0 < H2; k0 += 32) {
    v16bf a = frag_ld(arow, k0 + khalf);
    v16bf b = frag_ld(brow, k0 + khalf);
    acc = wmma_bf16(a, b, acc);
  }

  int n = r; // tag column
  if (n < Ktag) {
    float bias = h2t_b[n];
#pragma unroll
    for (int v = 0; v < 8; ++v) {
      int m = m0 + v + khalf;
      feats[(size_t)m * Ktag + n] = acc[v] + bias;
    }
  }
}

// ---------------- CRF Viterbi decode (one thread per batch row) ----------------
__global__ __launch_bounds__(128) void k_viterbi(const float* __restrict__ feats,
                                                 const int* __restrict__ tags,
                                                 const float* __restrict__ trans,
                                                 int* __restrict__ bps,
                                                 float* __restrict__ out) {
  __shared__ float trS[Ktag * Ktag];
  for (int i = threadIdx.x; i < Ktag * Ktag; i += 128) trS[i] = trans[i];
  __syncthreads();

  int b = threadIdx.x;
  const float* fb = feats + (size_t)b * Tlen * Ktag;

  float d[Ktag];
#pragma unroll
  for (int j = 0; j < Ktag; ++j) d[j] = fb[j];

  for (int t = 1; t < Tlen; ++t) {
    bool mt = tags[b * Tlen + t] > 0;
    float nd[Ktag];
    int bp[Ktag];
#pragma unroll
    for (int j = 0; j < Ktag; ++j) {
      float best = d[0] + trS[0 * Ktag + j];
      int bi = 0;
#pragma unroll
      for (int i = 1; i < Ktag; ++i) {
        float v = d[i] + trS[i * Ktag + j];
        if (v > best) { best = v; bi = i; }
      }
      nd[j] = best + fb[t * Ktag + j];
      bp[j] = bi;
    }
#pragma unroll
    for (int j = 0; j < Ktag; ++j) {
      d[j] = mt ? nd[j] : d[j];
      bps[((size_t)(t - 1) * Bsz + b) * Ktag + j] = mt ? bp[j] : j;
    }
  }

  float score = d[0];
  int last = 0;
#pragma unroll
  for (int j = 1; j < Ktag; ++j)
    if (d[j] > score) { score = d[j]; last = j; }

  out[b] = score;
  float* seq = out + Bsz + (size_t)b * Tlen;
  int carry = last;
  for (int s2 = Tlen - 2; s2 >= 0; --s2) {
    seq[1 + s2] = (float)carry;
    carry = bps[((size_t)s2 * Bsz + b) * Ktag + carry];
  }
  seq[0] = (float)carry;
}

// ---------------- host launch ----------------
extern "C" void kernel_launch(void* const* d_in, const int* in_sizes, int n_in,
                              void* d_out, int out_size, void* d_ws, size_t ws_size,
                              hipStream_t stream) {
  const int*   sentences = (const int*)  d_in[0];
  const int*   tags      = (const int*)  d_in[1];
  const float* embed     = (const float*)d_in[2];
  const float* w_ih_f    = (const float*)d_in[3];
  const float* w_hh_f    = (const float*)d_in[4];
  const float* b_ih_f    = (const float*)d_in[5];
  const float* b_hh_f    = (const float*)d_in[6];
  const float* w_ih_b    = (const float*)d_in[7];
  const float* w_hh_b    = (const float*)d_in[8];
  const float* b_ih_b    = (const float*)d_in[9];
  const float* b_hh_b    = (const float*)d_in[10];
  const float* h2t_w     = (const float*)d_in[11];
  const float* h2t_b     = (const float*)d_in[12];
  const float* trans     = (const float*)d_in[13];
  float* out = (float*)d_out;

  char* ws = (char*)d_ws;
  size_t off = 0;
  auto carve = [&](size_t bytes) {
    char* p = ws + off;
    off = (off + bytes + 255) & ~(size_t)255;
    return p;
  };
  __bf16* xBF     = (__bf16*)carve((size_t)MTOT * EPad * 2);             // 21 MB
  __bf16* wihBF   = (__bf16*)carve((size_t)2 * G4 * EPad * 2);           // 1.3 MB
  __bf16* whhBF   = (__bf16*)carve((size_t)2 * G4 * Hdim * 2);           // 1 MB
  __bf16* h2tBF   = (__bf16*)carve((size_t)16 * H2 * 2);                 // 16 KB
  __bf16* gatesIn = (__bf16*)carve((size_t)2 * MTOT * G4 * 2);           // 134 MB
  __bf16* lstmOut = (__bf16*)carve((size_t)MTOT * H2 * 2);               // 33.5 MB
  float*  feats   = (float*) carve((size_t)MTOT * Ktag * 4);             // 1.5 MB
  int*    bps     = (int*)   carve((size_t)(Tlen - 1) * Bsz * Ktag * 4); // 1.5 MB
  (void)ws_size; (void)in_sizes; (void)n_in; (void)out_size;

  // 1) weight/embedding prep
  {
    int total = 2 * G4 * EPad;
    k_prep_wih<<<(total + 255) / 256, 256, 0, stream>>>(w_ih_f, w_ih_b, wihBF);
  }
  {
    int total = 2 * G4 * Hdim;
    k_prep_whh<<<(total + 255) / 256, 256, 0, stream>>>(w_hh_f, w_hh_b, whhBF);
  }
  {
    int total = 16 * H2;
    k_prep_h2t<<<(total + 255) / 256, 256, 0, stream>>>(h2t_w, h2tBF);
  }
  {
    int total = MTOT * EPad;
    k_gather<<<(total + 255) / 256, 256, 0, stream>>>(sentences, embed, xBF);
  }

  // 2) batched input projection GEMM (WMMA bf16)
  {
    dim3 grid(MTOT / 16, (2 * G4) / 256);
    k_ingemm<<<grid, 128, 0, stream>>>(xBF, wihBF, gatesIn);
  }

  // 3) bidirectional LSTM recurrence (persistent WGs, WMMA + async LDS DMA)
  {
    dim3 grid(Bsz / 16, 2);
    k_lstm<<<grid, 512, 0, stream>>>(gatesIn, whhBF, b_ih_f, b_hh_f, b_ih_b, b_hh_b,
                                     lstmOut);
  }

  // 4) tag projection GEMM (WMMA bf16, N padded 12->16)
  k_featgemm<<<MTOT / 64, 128, 0, stream>>>(lstmOut, h2tBF, h2t_b, feats);

  // 5) CRF Viterbi decode + backtrack
  k_viterbi<<<1, 128, 0, stream>>>(feats, tags, trans, bps, out);
}
